// Model_70282844831811
// MI455X (gfx1250) — matrix operand order repaired and checked by reading, not compiled, not run on hardware
//
#include <hip/hip_runtime.h>
#include <hip/hip_bf16.h>
#include <math.h>

#define Bn 32
#define Tn 128
#define Dn 256
#define Hn 256
#define Vn 32000
#define SOS_TOK 126

typedef __bf16 bf_t;
typedef __attribute__((ext_vector_type(16))) __bf16 v16bf;
typedef __attribute__((ext_vector_type(8)))  __bf16 v8bf;
typedef __attribute__((ext_vector_type(8)))  float  v8f;

static __device__ __forceinline__ float sigmoidf_(float x) { return 1.0f / (1.0f + expf(-x)); }

// ---------------- utility kernels ----------------
__global__ void k_f2bf(const float* __restrict__ s, bf_t* __restrict__ d, int n) {
    int i = blockIdx.x * blockDim.x + threadIdx.x;
    if (i < n) d[i] = (bf_t)s[i];
}

__global__ void k_zero32(unsigned int* __restrict__ p, int n) {
    int i = blockIdx.x * blockDim.x + threadIdx.x;
    if (i < n) p[i] = 0u;
}

__global__ void k_enc_embed(const int* __restrict__ x, const float* __restrict__ emb,
                            bf_t* __restrict__ e_bf, int t) {
    int i = blockIdx.x * blockDim.x + threadIdx.x;
    if (i >= Bn * Dn) return;
    int b = i / Dn, d = i % Dn;
    int tok = x[b * Tn + t];
    e_bf[i] = (bf_t)emb[(size_t)tok * Dn + d];
}

// ---------------- WMMA fragment loaders (CDNA5 16x16x32 bf16 layouts, ISA 7.12.2) ----
// A (16x32, 16-bit): lane L: M = L%16, half = L/16; elems 0..7 -> K = 8*half+0..7,
// elems 8..15 -> K = 16+8*half+0..7  (two 16B contiguous loads).
static __device__ __forceinline__ v16bf load_a(const bf_t* __restrict__ A, int K, int m0, int k0, int lane) {
    int m = m0 + (lane & 15);
    int h = lane >> 4;
    const bf_t* row = A + (size_t)m * K + k0;
    v8bf lo = *(const v8bf*)(row + 8 * h);
    v8bf hi = *(const v8bf*)(row + 16 + 8 * h);
    v16bf r;
#pragma unroll
    for (int i = 0; i < 8; i++) { r[i] = lo[i]; r[8 + i] = hi[i]; }
    return r;
}
// B (32x16, 16-bit): lane L: N = L%16, hh = L/16; elems 0..7 -> K = 8*hh+0..7, elems 8..15 -> K = 16+8*hh+0..7.
// B[k][n] = W[n][k] since weights are (N,K) row-major and we compute X @ W^T.
static __device__ __forceinline__ v16bf load_b(const bf_t* __restrict__ W, int K, int n0, int k0, int lane) {
    int n = n0 + (lane & 15);
    int hh = lane >> 4;
    const bf_t* row = W + (size_t)n * K + k0 + 8 * hh;
    v8bf lo = *(const v8bf*)(row);
    v8bf hi = *(const v8bf*)(row + 16);
    v16bf r;
#pragma unroll
    for (int i = 0; i < 8; i++) { r[i] = lo[i]; r[8 + i] = hi[i]; }
    return r;
}

// Z[M x N] = Xa[M x Ka] @ Wa[N x Ka]^T (+ Xb[M x Kb] @ Wb[N x Kb]^T) (+ bias1 + bias2)
// grid: (N/64, M/16), block = 128 (4 waves); wave w owns n-tile (blockIdx.x*4 + w).
__global__ void k_gemm(const bf_t* __restrict__ Xa, const bf_t* __restrict__ Wa, int Ka,
                       const bf_t* __restrict__ Xb, const bf_t* __restrict__ Wb, int Kb,
                       const float* __restrict__ bias1, const float* __restrict__ bias2,
                       float* __restrict__ Z, int N) {
    int lane = threadIdx.x & 31;
    int wave = threadIdx.x >> 5;
    int n0 = (blockIdx.x * 4 + wave) * 16;
    int m0 = blockIdx.y * 16;
    v8f c = {};
    for (int k0 = 0; k0 < Ka; k0 += 32) {
        v16bf a = load_a(Xa, Ka, m0, k0, lane);
        v16bf b = load_b(Wa, Ka, n0, k0, lane);
        c = __builtin_amdgcn_wmma_f32_16x16x32_bf16(false, a, false, b, (short)0, c, false, false);
    }
    if (Xb) {
        for (int k0 = 0; k0 < Kb; k0 += 32) {
            v16bf a = load_a(Xb, Kb, m0, k0, lane);
            v16bf b = load_b(Wb, Kb, n0, k0, lane);
            c = __builtin_amdgcn_wmma_f32_16x16x32_bf16(false, a, false, b, (short)0, c, false, false);
        }
    }
    int n = n0 + (lane & 15);
    int half = lane >> 4;
    float badd = 0.f;
    if (bias1) badd += bias1[n];
    if (bias2) badd += bias2[n];
#pragma unroll
    for (int r = 0; r < 8; r++) {       // C/D layout: VGPR r -> M = r + 8*half, N = lane%16
        int m = m0 + r + 8 * half;
        Z[(size_t)m * N + n] = c[r] + badd;
    }
}

// ---------------- LSTM pointwise cell (gate order i,f,g,o) ----------------
__global__ void k_lstm_cell(const float* __restrict__ Z, float* __restrict__ h, float* __restrict__ c,
                            bf_t* __restrict__ hbf, float* __restrict__ eo, bf_t* __restrict__ eobf,
                            int Hd, int t, int off) {
    int i = blockIdx.x * blockDim.x + threadIdx.x;
    if (i >= Bn * Hd) return;
    int b = i / Hd, j = i % Hd;
    const float* z = Z + (size_t)b * 4 * Hd;
    float ig = sigmoidf_(z[j]);
    float fg = sigmoidf_(z[Hd + j]);
    float gg = tanhf(z[2 * Hd + j]);
    float og = sigmoidf_(z[3 * Hd + j]);
    float c2 = fg * c[i] + ig * gg;
    float h2 = og * tanhf(c2);
    c[i] = c2; h[i] = h2; hbf[i] = (bf_t)h2;
    if (eo) {
        size_t o = ((size_t)b * Tn + t) * (2 * Hn) + off + j;
        eo[o] = h2; eobf[o] = (bf_t)h2;
    }
}

__global__ void k_dec_init(const float* __restrict__ hf, const float* __restrict__ hb,
                           const float* __restrict__ cf, const float* __restrict__ cb,
                           float* __restrict__ h, float* __restrict__ c,
                           bf_t* __restrict__ hbf, int* __restrict__ tok) {
    int i = blockIdx.x * blockDim.x + threadIdx.x;
    if (i < Bn) tok[i] = SOS_TOK;
    if (i >= Bn * 2 * Hn) return;
    int b = i / (2 * Hn), j = i % (2 * Hn);
    float hv = (j < Hn) ? hf[b * Hn + j] : hb[b * Hn + (j - Hn)];
    float cv = (j < Hn) ? cf[b * Hn + j] : cb[b * Hn + (j - Hn)];
    h[i] = hv; c[i] = cv; hbf[i] = (bf_t)hv;
}

__global__ void k_dec_embed(const int* __restrict__ tok, const float* __restrict__ emb,
                            bf_t* __restrict__ xcat) {
    int i = blockIdx.x * blockDim.x + threadIdx.x;
    if (i >= Bn * Dn) return;
    int b = i / Dn, d = i % Dn;
    xcat[(size_t)b * (Dn + 2 * Hn) + d] = (bf_t)emb[(size_t)tok[b] * Dn + d];
}

__global__ void k_attn_energy(const float* __restrict__ pe, const float* __restrict__ q,
                              const float* __restrict__ ba, const float* __restrict__ va,
                              float* __restrict__ energy) {
    int i = blockIdx.x * blockDim.x + threadIdx.x;
    if (i >= Bn * Tn) return;
    int b = i / Tn;
    const float* p = pe + (size_t)i * (2 * Hn);
    const float* qq = q + (size_t)b * (2 * Hn);
    float s = 0.f;
    for (int j = 0; j < 2 * Hn; j++) s += tanhf(p[j] + qq[j] + ba[j]) * va[j];
    energy[i] = s;
}

// one block (Tn threads) per batch row: softmax over T + weighted sum of enc_out -> xcat ctx half
__global__ void k_attn_ctx(const float* __restrict__ energy, const float* __restrict__ eo,
                           bf_t* __restrict__ xcat) {
    __shared__ float red[Tn];
    __shared__ float w[Tn];
    int b = blockIdx.x;
    int t = threadIdx.x;
    float e = energy[b * Tn + t];
    red[t] = e; __syncthreads();
    for (int s = Tn / 2; s > 0; s >>= 1) { if (t < s) red[t] = fmaxf(red[t], red[t + s]); __syncthreads(); }
    float mx = red[0]; __syncthreads();
    float ex = expf(e - mx);
    red[t] = ex; __syncthreads();
    for (int s = Tn / 2; s > 0; s >>= 1) { if (t < s) red[t] += red[t + s]; __syncthreads(); }
    float inv = 1.0f / red[0];
    w[t] = ex * inv; __syncthreads();
    for (int j = t; j < 2 * Hn; j += Tn) {
        float s2 = 0.f;
        for (int tt = 0; tt < Tn; tt++) s2 += w[tt] * eo[((size_t)b * Tn + tt) * (2 * Hn) + j];
        xcat[(size_t)b * (Dn + 2 * Hn) + Dn + j] = (bf_t)s2;
    }
}

// one block (256 threads) per batch row: log_softmax over V + first-index argmax -> tok
__global__ void k_out(const float* __restrict__ logits, float* __restrict__ out,
                      int* __restrict__ tok, int t) {
    __shared__ float rv[256];
    __shared__ int   ri[256];
    int b = blockIdx.x;
    int tid = threadIdx.x;
    const float* lg = logits + (size_t)b * Vn;
    float mx = -3.402823466e+38f; int mi = 0;
    for (int v = tid; v < Vn; v += 256) { float f = lg[v]; if (f > mx) { mx = f; mi = v; } }
    rv[tid] = mx; ri[tid] = mi; __syncthreads();
    for (int s = 128; s > 0; s >>= 1) {
        if (tid < s) {
            float a = rv[tid], bb = rv[tid + s];
            if (bb > a || (bb == a && ri[tid + s] < ri[tid])) { rv[tid] = bb; ri[tid] = ri[tid + s]; }
        }
        __syncthreads();
    }
    float MX = rv[0]; int MI = ri[0]; __syncthreads();
    float sum = 0.f;
    for (int v = tid; v < Vn; v += 256) sum += expf(lg[v] - MX);
    rv[tid] = sum; __syncthreads();
    for (int s = 128; s > 0; s >>= 1) { if (tid < s) rv[tid] += rv[tid + s]; __syncthreads(); }
    float lse = MX + logf(rv[0]);
    float* o = out + ((size_t)b * Tn + t) * Vn;
    for (int v = tid; v < Vn; v += 256) o[v] = lg[v] - lse;
    if (tid == 0) tok[b] = MI;
}

// ---------------- host orchestration ----------------
extern "C" void kernel_launch(void* const* d_in, const int* in_sizes, int n_in,
                              void* d_out, int out_size, void* d_ws, size_t ws_size,
                              hipStream_t stream) {
    (void)in_sizes; (void)n_in; (void)out_size; (void)ws_size;
    const int*   x       = (const int*)  d_in[0];
    const float* emb_enc = (const float*)d_in[1];
    const float* Wih_f   = (const float*)d_in[2];
    const float* Whh_f   = (const float*)d_in[3];
    const float* bih_f   = (const float*)d_in[4];
    const float* bhh_f   = (const float*)d_in[5];
    const float* Wih_b   = (const float*)d_in[6];
    const float* Whh_b   = (const float*)d_in[7];
    const float* bih_b   = (const float*)d_in[8];
    const float* bhh_b   = (const float*)d_in[9];
    const float* emb_dec = (const float*)d_in[10];
    const float* Wa_h    = (const float*)d_in[11];
    const float* Wa_e    = (const float*)d_in[12];
    const float* v_a     = (const float*)d_in[13];
    const float* b_a     = (const float*)d_in[14];
    const float* Wih_d   = (const float*)d_in[15];
    const float* Whh_d   = (const float*)d_in[16];
    const float* bih_d   = (const float*)d_in[17];
    const float* bhh_d   = (const float*)d_in[18];
    const float* Wout    = (const float*)d_in[19];
    const float* bout    = (const float*)d_in[20];
    float* out = (float*)d_out;

    char* base = (char*)d_ws;
    size_t off = 0;
    auto alloc = [&](size_t bytes) -> char* {
        char* p = base + off;
        off += (bytes + 255) & ~((size_t)255);
        return p;
    };

    // bf16 weight copies
    bf_t* wih_f_bf = (bf_t*)alloc((size_t)4 * Hn * Dn * 2);
    bf_t* whh_f_bf = (bf_t*)alloc((size_t)4 * Hn * Hn * 2);
    bf_t* wih_b_bf = (bf_t*)alloc((size_t)4 * Hn * Dn * 2);
    bf_t* whh_b_bf = (bf_t*)alloc((size_t)4 * Hn * Hn * 2);
    bf_t* wa_e_bf  = (bf_t*)alloc((size_t)2 * Hn * 2 * Hn * 2);
    bf_t* wa_h_bf  = (bf_t*)alloc((size_t)2 * Hn * 2 * Hn * 2);
    bf_t* wih_d_bf = (bf_t*)alloc((size_t)8 * Hn * (Dn + 2 * Hn) * 2);
    bf_t* whh_d_bf = (bf_t*)alloc((size_t)8 * Hn * 2 * Hn * 2);
    bf_t* wout_bf  = (bf_t*)alloc((size_t)Vn * 2 * Hn * 2);
    // activations
    bf_t*  e_bf   = (bf_t*)alloc((size_t)Bn * Dn * 2);
    float* hF     = (float*)alloc((size_t)Bn * Hn * 4);
    float* cF     = (float*)alloc((size_t)Bn * Hn * 4);
    float* hB     = (float*)alloc((size_t)Bn * Hn * 4);
    float* cB     = (float*)alloc((size_t)Bn * Hn * 4);
    bf_t*  hF_bf  = (bf_t*)alloc((size_t)Bn * Hn * 2);
    bf_t*  hB_bf  = (bf_t*)alloc((size_t)Bn * Hn * 2);
    float* zF     = (float*)alloc((size_t)Bn * 4 * Hn * 4);
    float* zB     = (float*)alloc((size_t)Bn * 4 * Hn * 4);
    float* enc_out    = (float*)alloc((size_t)Bn * Tn * 2 * Hn * 4);
    bf_t*  enc_out_bf = (bf_t*)alloc((size_t)Bn * Tn * 2 * Hn * 2);
    float* proj_e = (float*)alloc((size_t)Bn * Tn * 2 * Hn * 4);
    float* h_d    = (float*)alloc((size_t)Bn * 2 * Hn * 4);
    float* c_d    = (float*)alloc((size_t)Bn * 2 * Hn * 4);
    bf_t*  h_d_bf = (bf_t*)alloc((size_t)Bn * 2 * Hn * 2);
    float* q      = (float*)alloc((size_t)Bn * 2 * Hn * 4);
    float* energy = (float*)alloc((size_t)Bn * Tn * 4);
    bf_t*  xcat   = (bf_t*)alloc((size_t)Bn * (Dn + 2 * Hn) * 2);
    float* z_d    = (float*)alloc((size_t)Bn * 8 * Hn * 4);
    float* logits = (float*)alloc((size_t)Bn * Vn * 4);
    int*   tok    = (int*)alloc((size_t)Bn * 4);

    auto conv = [&](const float* s, bf_t* dd, int n) {
        k_f2bf<<<dim3((n + 255) / 256), dim3(256), 0, stream>>>(s, dd, n);
    };
    conv(Wih_f, wih_f_bf, 4 * Hn * Dn);
    conv(Whh_f, whh_f_bf, 4 * Hn * Hn);
    conv(Wih_b, wih_b_bf, 4 * Hn * Dn);
    conv(Whh_b, whh_b_bf, 4 * Hn * Hn);
    conv(Wa_e,  wa_e_bf,  2 * Hn * 2 * Hn);
    conv(Wa_h,  wa_h_bf,  2 * Hn * 2 * Hn);
    conv(Wih_d, wih_d_bf, 8 * Hn * (Dn + 2 * Hn));
    conv(Whh_d, whh_d_bf, 8 * Hn * 2 * Hn);
    conv(Wout,  wout_bf,  Vn * 2 * Hn);

    auto zero = [&](void* p, size_t words) {
        k_zero32<<<dim3((unsigned)((words + 255) / 256)), dim3(256), 0, stream>>>((unsigned int*)p, (int)words);
    };
    zero(hF, Bn * Hn); zero(cF, Bn * Hn); zero(hB, Bn * Hn); zero(cB, Bn * Hn);
    zero(hF_bf, Bn * Hn / 2); zero(hB_bf, Bn * Hn / 2);

    // ---- encoder (both directions stepped forward in time, as in the reference) ----
    dim3 gEnc(4 * Hn / 64, Bn / 16);   // N=1024: (16,2)
    for (int t = 0; t < Tn; t++) {
        k_enc_embed<<<dim3((Bn * Dn + 255) / 256), dim3(256), 0, stream>>>(x, emb_enc, e_bf, t);
        k_gemm<<<gEnc, 128, 0, stream>>>(e_bf, wih_f_bf, Dn, hF_bf, whh_f_bf, Hn, bih_f, bhh_f, zF, 4 * Hn);
        k_gemm<<<gEnc, 128, 0, stream>>>(e_bf, wih_b_bf, Dn, hB_bf, whh_b_bf, Hn, bih_b, bhh_b, zB, 4 * Hn);
        k_lstm_cell<<<dim3((Bn * Hn + 255) / 256), dim3(256), 0, stream>>>(zF, hF, cF, hF_bf, enc_out, enc_out_bf, Hn, t, 0);
        k_lstm_cell<<<dim3((Bn * Hn + 255) / 256), dim3(256), 0, stream>>>(zB, hB, cB, hB_bf, enc_out, enc_out_bf, Hn, t, Hn);
    }

    // ---- attention key projection: proj_e = enc_out @ Wa_e^T  (M = B*T = 4096) ----
    k_gemm<<<dim3(2 * Hn / 64, (Bn * Tn) / 16), 128, 0, stream>>>(
        enc_out_bf, wa_e_bf, 2 * Hn, nullptr, nullptr, 0, nullptr, nullptr, proj_e, 2 * Hn);

    // ---- decoder init ----
    k_dec_init<<<dim3((Bn * 2 * Hn + 255) / 256), dim3(256), 0, stream>>>(hF, hB, cF, cB, h_d, c_d, h_d_bf, tok);

    // ---- decoder loop ----
    dim3 gQ(2 * Hn / 64, Bn / 16);       // (8,2)
    dim3 gD(8 * Hn / 64, Bn / 16);       // (32,2)
    dim3 gO(Vn / 64, Bn / 16);           // (500,2)
    for (int t = 0; t < Tn; t++) {
        k_dec_embed<<<dim3((Bn * Dn + 255) / 256), dim3(256), 0, stream>>>(tok, emb_dec, xcat);
        k_gemm<<<gQ, 128, 0, stream>>>(h_d_bf, wa_h_bf, 2 * Hn, nullptr, nullptr, 0, nullptr, nullptr, q, 2 * Hn);
        k_attn_energy<<<dim3((Bn * Tn + 255) / 256), dim3(256), 0, stream>>>(proj_e, q, b_a, v_a, energy);
        k_attn_ctx<<<dim3(Bn), dim3(Tn), 0, stream>>>(energy, enc_out, xcat);
        k_gemm<<<gD, 128, 0, stream>>>(xcat, wih_d_bf, Dn + 2 * Hn, h_d_bf, whh_d_bf, 2 * Hn, bih_d, bhh_d, z_d, 8 * Hn);
        k_lstm_cell<<<dim3((Bn * 2 * Hn + 255) / 256), dim3(256), 0, stream>>>(z_d, h_d, c_d, h_d_bf, nullptr, nullptr, 2 * Hn, 0, 0);
        k_gemm<<<gO, 128, 0, stream>>>(h_d_bf, wout_bf, 2 * Hn, nullptr, nullptr, 0, bout, nullptr, logits, Vn);
        k_out<<<dim3(Bn), dim3(256), 0, stream>>>(logits, out, tok, t);
    }
}
